// ModernBERTAttention_30648886624595
// MI455X (gfx1250) — matrix operand order repaired and checked by reading, not compiled, not run on hardware
//
#include <hip/hip_runtime.h>

typedef _Float16 v8h  __attribute__((ext_vector_type(8)));
typedef _Float16 v16h __attribute__((ext_vector_type(16)));
typedef float    v8f  __attribute__((ext_vector_type(8)));

constexpr int Bn   = 2;
constexpr int Sn   = 2048;
constexpr int Hn   = 768;
constexpr int NH   = 12;
constexpr int HD   = 64;
constexpr int QKVN = 3 * Hn;        // 2304
constexpr int Mtot = Bn * Sn;       // 4096

__device__ __forceinline__ v16h combine16(v8h lo, v8h hi) {
  return __builtin_shufflevector(lo, hi, 0,1,2,3,4,5,6,7,8,9,10,11,12,13,14,15);
}

// A fragment, 16x32 f16, source row-major [M][ld]
__device__ __forceinline__ v16h load_afrag(const _Float16* base, int ld, int row0, int k0) {
  int lane = threadIdx.x & 31;
  int half = lane >> 4, lr = lane & 15;
  const _Float16* p = base + (size_t)(row0 + lr) * ld + k0 + half * 8;
  return combine16(*(const v8h*)p, *(const v8h*)(p + 16));
}

// B fragment, 32x16 f16, source K-major [K][ld] (N contiguous)
__device__ __forceinline__ v16h load_bfrag(const _Float16* base, int ld, int k0, int n0) {
  int lane = threadIdx.x & 31;
  int half = lane >> 4, lr = lane & 15;
  v16h r;
#pragma unroll
  for (int s = 0; s < 16; s++) {
    int k = k0 + (s < 8 ? half * 8 + s : 16 + half * 8 + (s - 8));
    r[s] = base[(size_t)k * ld + n0 + lr];
  }
  return r;
}

__device__ __forceinline__ v8f wmma16(v16h a, v16h b, v8f c) {
  return __builtin_amdgcn_wmma_f32_16x16x32_f16(false, a, false, b, (short)0, c, false, false);
}

// ---------------- conversion kernels ----------------
__global__ void mba_cvt_f16(const float* __restrict__ in, _Float16* __restrict__ out, int n) {
  int i = blockIdx.x * 256 + threadIdx.x;
  if (i < n) out[i] = (_Float16)in[i];
}

__global__ void mba_cvt_tr_f16(const float* __restrict__ in, _Float16* __restrict__ out,
                               int rows, int cols) {
  int i = blockIdx.x * 256 + threadIdx.x;
  if (i < rows * cols) {
    int r = i / cols, c = i - r * cols;
    out[(size_t)c * rows + r] = (_Float16)in[i];
  }
}

// ---------------- QKV GEMM + RoPE epilogue ----------------
// grid: (QKVN/256, Mtot/16), block 128 (4 waves); wave tile 16(m) x 64(n)
__global__ __launch_bounds__(128)
void mba_qkv_rope(const _Float16* __restrict__ Xh, const _Float16* __restrict__ WqkvT,
                  const int* __restrict__ posids,
                  _Float16* __restrict__ Qh, _Float16* __restrict__ Kt,
                  _Float16* __restrict__ Vh) {
  int wave = threadIdx.x >> 5, lane = threadIdx.x & 31;
  int half = lane >> 4, lr = lane & 15;
  int n0 = blockIdx.x * 256 + wave * 64;
  int m0 = blockIdx.y * 16;

  v8f acc[4] = {};
  for (int k0 = 0; k0 < Hn; k0 += 32) {
    v16h a = load_afrag(Xh, Hn, m0, k0);
#pragma unroll
    for (int t = 0; t < 4; t++) {
      v16h b = load_bfrag(WqkvT, QKVN, k0, n0 + t * 16);
      acc[t] = wmma16(a, b, acc[t]);
    }
  }

#pragma unroll
  for (int t = 0; t < 4; t++) {
    int f0 = n0 + t * 16;
    int which = f0 / Hn;             // 0=q 1=k 2=v (uniform per wave-subtile)
    int h = (f0 % Hn) / HD;
    int d = (f0 % HD) + lr;          // this lane's head-dim index
    if (which == 2) {
#pragma unroll
      for (int r = 0; r < 8; r++) {
        int m = m0 + r + half * 8;
        int b = m >> 11, s = m & 2047;
        Vh[(size_t)((b * NH + h) * Sn + s) * HD + d] = (_Float16)acc[t][r];
      }
    } else {
      int j = d >> 1;
      float freq = __powf(10000.0f, -(float)(j & 15) * (1.0f / 16.0f));
#pragma unroll
      for (int r = 0; r < 8; r++) {
        float own = acc[t][r];
        float other = __shfl_xor(own, 1, 32);   // rope partner column
        int m = m0 + r + half * 8;
        int b = m >> 11, s = m & 2047;
        float pos = (float)posids[m];
        float sn, cs;
        __sincosf(pos * freq, &sn, &cs);
        float out = (d & 1) ? (own * cs + other * sn) : (own * cs - other * sn);
        if (which == 0) {
          out *= 0.125f;   // HEAD_DIM^-0.5
          Qh[(size_t)((b * NH + h) * Sn + s) * HD + d] = (_Float16)out;
        } else {
          Kt[(size_t)((b * NH + h) * HD + d) * Sn + s] = (_Float16)out;  // transposed
        }
      }
    }
  }
}

// ---------------- flash attention ----------------
// grid: 24*16 blocks, 256 threads (8 waves); wave = 16 queries, tiles of 32 keys
__global__ __launch_bounds__(256)
void mba_flash(const _Float16* __restrict__ Qh, const _Float16* __restrict__ Kt,
               const _Float16* __restrict__ Vh, _Float16* __restrict__ Ctx) {
  __shared__ __align__(16) _Float16 plds[8][512];   // per-wave 16x32 P tile

  int wave = threadIdx.x >> 5, lane = threadIdx.x & 31;
  int half = lane >> 4, lr = lane & 15;
  int bh = blockIdx.x >> 4;
  int q0 = (blockIdx.x & 15) * 128 + wave * 16;

  const _Float16* Qb  = Qh + (size_t)bh * Sn * HD;
  const _Float16* Ktb = Kt + (size_t)bh * HD * Sn;
  const _Float16* Vb  = Vh + (size_t)bh * Sn * HD;
  _Float16* pw = &plds[wave][0];

  v16h a0 = load_afrag(Qb, HD, q0, 0);
  v16h a1 = load_afrag(Qb, HD, q0, 32);

  float mrun[8], lrun[8];
#pragma unroll
  for (int r = 0; r < 8; r++) { mrun[r] = -1e30f; lrun[r] = 0.0f; }
  v8f o[4] = {};

  for (int kt = 0; kt < Sn; kt += 32) {
    v16h b00 = load_bfrag(Ktb, Sn, 0,  kt);       // d 0..31, keys kt..kt+15
    v16h b01 = load_bfrag(Ktb, Sn, 32, kt);
    v16h b10 = load_bfrag(Ktb, Sn, 0,  kt + 16);
    v16h b11 = load_bfrag(Ktb, Sn, 32, kt + 16);
    v8f c0 = {}; c0 = wmma16(a0, b00, c0); c0 = wmma16(a1, b01, c0);
    v8f c1 = {}; c1 = wmma16(a0, b10, c1); c1 = wmma16(a1, b11, c1);

    float corr[8];
#pragma unroll
    for (int r = 0; r < 8; r++) {
      float mx = fmaxf(c0[r], c1[r]);
      mx = fmaxf(mx, __shfl_xor(mx, 1, 32));
      mx = fmaxf(mx, __shfl_xor(mx, 2, 32));
      mx = fmaxf(mx, __shfl_xor(mx, 4, 32));
      mx = fmaxf(mx, __shfl_xor(mx, 8, 32));
      float mnew = fmaxf(mrun[r], mx);
      float p0 = __expf(c0[r] - mnew);
      float p1 = __expf(c1[r] - mnew);
      float cf = __expf(mrun[r] - mnew);
      mrun[r] = mnew;
      float rs = p0 + p1;
      rs += __shfl_xor(rs, 1, 32);
      rs += __shfl_xor(rs, 2, 32);
      rs += __shfl_xor(rs, 4, 32);
      rs += __shfl_xor(rs, 8, 32);
      lrun[r] = lrun[r] * cf + rs;
      corr[r] = cf;
      // stage P (16q x 32k) into LDS, packed 2 keys/dword by even lanes
      float p0o = __shfl_xor(p0, 1, 32);
      float p1o = __shfl_xor(p1, 1, 32);
      if (!(lane & 1)) {
        int q = r + half * 8;
        union { unsigned u; _Float16 h2[2]; } w0, w1;
        w0.h2[0] = (_Float16)p0; w0.h2[1] = (_Float16)p0o;
        w1.h2[0] = (_Float16)p1; w1.h2[1] = (_Float16)p1o;
        *(unsigned*)(pw + q * 32 + lr)      = w0.u;
        *(unsigned*)(pw + q * 32 + 16 + lr) = w1.u;
      }
    }
    // wave-private LDS RAW fence (split DS counter, CDNA5)
    asm volatile("s_wait_dscnt 0x0" ::: "memory");

    v16h pA = load_afrag(pw, 32, 0, 0);
#pragma unroll
    for (int dn = 0; dn < 4; dn++) {
      v16h bv = load_bfrag(Vb, HD, kt, dn * 16);
#pragma unroll
      for (int r = 0; r < 8; r++) o[dn][r] *= corr[r];
      o[dn] = wmma16(pA, bv, o[dn]);
    }
  }

  int b = bh / NH, h = bh % NH;
#pragma unroll
  for (int r = 0; r < 8; r++) {
    float inv = 1.0f / lrun[r];
    int s = q0 + r + half * 8;
    size_t rowoff = ((size_t)(b * Sn + s)) * Hn + h * HD;
#pragma unroll
    for (int dn = 0; dn < 4; dn++)
      Ctx[rowoff + dn * 16 + lr] = (_Float16)(o[dn][r] * inv);
  }
}

// ---------------- output projection ----------------
// grid: (Hn/256, Mtot/16), block 128; wave tile 16 x 64, fp32 out
__global__ __launch_bounds__(128)
void mba_outproj(const _Float16* __restrict__ Ctx, const _Float16* __restrict__ WoT,
                 float* __restrict__ Out) {
  int wave = threadIdx.x >> 5, lane = threadIdx.x & 31;
  int half = lane >> 4, lr = lane & 15;
  int n0 = blockIdx.x * 256 + wave * 64;
  int m0 = blockIdx.y * 16;

  v8f acc[4] = {};
  for (int k0 = 0; k0 < Hn; k0 += 32) {
    v16h a = load_afrag(Ctx, Hn, m0, k0);
#pragma unroll
    for (int t = 0; t < 4; t++) {
      v16h b = load_bfrag(WoT, Hn, k0, n0 + t * 16);
      acc[t] = wmma16(a, b, acc[t]);
    }
  }
#pragma unroll
  for (int t = 0; t < 4; t++)
#pragma unroll
    for (int r = 0; r < 8; r++)
      Out[(size_t)(m0 + r + half * 8) * Hn + n0 + t * 16 + lr] = acc[t][r];
}

extern "C" void kernel_launch(void* const* d_in, const int* in_sizes, int n_in,
                              void* d_out, int out_size, void* d_ws, size_t ws_size,
                              hipStream_t stream) {
  (void)in_sizes; (void)n_in; (void)out_size; (void)ws_size;
  const float* hs   = (const float*)d_in[0];   // (2,2048,768)
  const float* Wqkv = (const float*)d_in[1];   // (2304,768)
  const float* Wo   = (const float*)d_in[2];   // (768,768)
  const int*   pid  = (const int*)d_in[3];     // (2,2048)

  _Float16* ws = (_Float16*)d_ws;
  size_t off = 0;
  _Float16* Xh    = ws + off; off += (size_t)Mtot * Hn;     // 4096*768
  _Float16* WqkvT = ws + off; off += (size_t)QKVN * Hn;     // 768*2304 (K-major)
  _Float16* WoT   = ws + off; off += (size_t)Hn * Hn;       // 768*768  (K-major)
  _Float16* Qh    = ws + off; off += (size_t)Bn * NH * Sn * HD;
  _Float16* Kt    = ws + off; off += (size_t)Bn * NH * HD * Sn;
  _Float16* Vh    = ws + off; off += (size_t)Bn * NH * Sn * HD;
  _Float16* Ctx   = ws + off;

  mba_cvt_f16<<<(Mtot * Hn + 255) / 256, 256, 0, stream>>>(hs, Xh, Mtot * Hn);
  mba_cvt_tr_f16<<<(QKVN * Hn + 255) / 256, 256, 0, stream>>>(Wqkv, WqkvT, QKVN, Hn);
  mba_cvt_tr_f16<<<(Hn * Hn + 255) / 256, 256, 0, stream>>>(Wo, WoT, Hn, Hn);

  mba_qkv_rope<<<dim3(QKVN / 256, Mtot / 16), 128, 0, stream>>>(Xh, WqkvT, pid, Qh, Kt, Vh);
  mba_flash<<<Bn * NH * (Sn / 128), 256, 0, stream>>>(Qh, Kt, Vh, Ctx);
  mba_outproj<<<dim3(Hn / 256, Mtot / 16), 128, 0, stream>>>(Ctx, WoT, (float*)d_out);
}
